// Model_33466385170973
// MI455X (gfx1250) — matrix operand order repaired and checked by reading, not compiled, not run on hardware
//
#include <hip/hip_runtime.h>
#include <hip/hip_bf16.h>

// ---------------------------------------------------------------------------
// Types for CDNA5 WMMA (wave32, 16x16x32 bf16 -> f32)
// ---------------------------------------------------------------------------
typedef __attribute__((ext_vector_type(16))) __bf16 v16bf;
typedef __attribute__((ext_vector_type(8)))  __bf16 v8bf;
typedef __attribute__((ext_vector_type(8)))  float  v8f;

#define NTREE   128
#define TNODES  255
#define NNODES  (NTREE * TNODES)   // 32640
#define DMODEL  256

__device__ __forceinline__ float sigmoidf_(float x) { return 1.0f / (1.0f + __expf(-x)); }

__device__ __forceinline__ v16bf load_frag16(const __bf16* p0, const __bf16* p1) {
    v8bf a = *(const v8bf*)p0;
    v8bf b = *(const v8bf*)p1;
    v16bf r;
#pragma unroll
    for (int i = 0; i < 8; ++i) { r[i] = a[i]; r[i + 8] = b[i]; }
    return r;
}

// ---------------------------------------------------------------------------
// Generic bf16 WMMA GEMM:  out[M x O] (f32) = gather(A)[M x K] @ W[O x K]^T
// Row gather: row_src = g1 ? (g2 ? g2[g1[row]] : g1[row]) : row
// One wave computes a 64x16 output strip (4 M-tiles x 1 O-tile), reusing the
// B fragment across the 4 WMMAs of each k-step.  K == 256 here (8 k-steps).
// ---------------------------------------------------------------------------
#define MREP 4
__global__ void wmma_gemm_kernel(const __bf16* __restrict__ A,
                                 const __bf16* __restrict__ W,
                                 float* __restrict__ out,
                                 int M, int O, int K,
                                 const int* __restrict__ g1,
                                 const int* __restrict__ g2)
{
    const int lane = threadIdx.x;            // 0..31
    const int half = lane >> 4;              // 0: lanes 0-15, 1: lanes 16-31
    const int lr   = lane & 15;
    const int ot   = blockIdx.x;             // O/16 tiles
    const int mw   = blockIdx.y * blockDim.y + threadIdx.y;   // 64-row strip id
    if (mw * (MREP * 16) >= M) return;

    const __bf16* arow[MREP];
#pragma unroll
    for (int r = 0; r < MREP; ++r) {
        int row = mw * (MREP * 16) + r * 16 + lr;
        int src = row;
        if (g1) { src = g1[row]; if (g2) src = g2[src]; }
        arow[r] = A + (size_t)src * K;
    }
    const __bf16* brow = W + (size_t)(ot * 16 + lr) * K;

    v8f acc[MREP];
#pragma unroll
    for (int r = 0; r < MREP; ++r)
#pragma unroll
        for (int q = 0; q < 8; ++q) acc[r][q] = 0.0f;

#pragma unroll
    for (int kb = 0; kb < 256; kb += 32) {
        // B fragment (32x16): lanes 0-15 hold K=kb..kb+15 of col lr,
        //                     lanes 16-31 hold K=kb+16..kb+31.
        const __bf16* bp = brow + kb + half * 16;
        v16bf bfrag = load_frag16(bp, bp + 8);
#pragma unroll
        for (int r = 0; r < MREP; ++r) {
            // A fragment (16x32): lane<16 -> K 0..15, lane>=16 -> K 16..31.
            const __bf16* ap = arow[r] + kb + half * 16;
            v16bf afrag = load_frag16(ap, ap + 8);
            acc[r] = __builtin_amdgcn_wmma_f32_16x16x32_bf16(
                false, afrag, false, bfrag, (short)0, acc[r], false, false);
        }
    }

    const int col = ot * 16 + lr;
#pragma unroll
    for (int r = 0; r < MREP; ++r) {
        int mbase = mw * (MREP * 16) + r * 16 + half * 8;  // VGPR q -> row mbase+q
        float* op = out + (size_t)mbase * O + col;
#pragma unroll
        for (int q = 0; q < 8; ++q) op[(size_t)q * O] = acc[r][q];
    }
}

// ---------------------------------------------------------------------------
// f32 -> bf16 conversion (weights / embedding staging)
// ---------------------------------------------------------------------------
__global__ void f32_to_bf16_kernel(const float* __restrict__ src,
                                   __bf16* __restrict__ dst, int n)
{
    int i = blockIdx.x * blockDim.x + threadIdx.x;
    if (i < n) dst[i] = (__bf16)src[i];
}

// ---------------------------------------------------------------------------
// Heap-structure index tables (run-independent).
//  act[]  : per-level compact index -> node id
//  child[]: per-level, per-parent two child node ids
// heap hp in [0,254]; node id = b*255 + (254-hp); depth = floor(log2(hp+1));
// level = 7-depth.  actOff(depth d) = 128*(256 - 2^(d+1));
// childOff(depth d) = 256*(128 - 2^(d+1)).
// ---------------------------------------------------------------------------
__global__ void build_idx_kernel(int* __restrict__ act, int* __restrict__ child)
{
    int i = blockIdx.x * blockDim.x + threadIdx.x;
    if (i >= NTREE * TNODES) return;
    int b  = i / TNODES;
    int hp = i % TNODES;
    int dep  = 31 - __builtin_clz((unsigned)(hp + 1));
    int j    = hp - ((1 << dep) - 1);
    int apos = NTREE * (256 - (1 << (dep + 1))) + b * (1 << dep) + j;
    act[apos] = b * TNODES + (TNODES - 1 - hp);
    if (hp < (TNODES - 1) / 2) {   // internal node -> has two children
        int cpos = 2 * NTREE * (128 - (1 << (dep + 1))) + (b * (1 << dep) + j) * 2;
        child[cpos]     = b * TNODES + (TNODES - 1 - (2 * hp + 1));
        child[cpos + 1] = b * TNODES + (TNODES - 1 - (2 * hp + 2));
    }
}

// ---------------------------------------------------------------------------
// Edge step: f = sigmoid(x_f[parent] + b_f + U_f h_child); accumulate
// c_sum = f0*c0 + f1*c1 and h_sum = h0 + h1 (bf16 for next WMMA GEMM).
// ---------------------------------------------------------------------------
__global__ void edge_kernel(const float* __restrict__ fraw,   // (2*nact) x 256
                            const float* __restrict__ xf,     // nact x 256
                            const float* __restrict__ bf,     // 256
                            const float* __restrict__ h,
                            const float* __restrict__ c,
                            const int*   __restrict__ child,  // 2*nact
                            float* __restrict__ csum,         // nact x 256
                            __bf16* __restrict__ hsum,        // nact x 256
                            int nact)
{
    int idx = blockIdx.x * blockDim.x + threadIdx.x;
    if (idx >= nact * DMODEL) return;
    int p = idx >> 8, j = idx & 255;
    int c0 = child[2 * p], c1 = child[2 * p + 1];
    float xfb = xf[(size_t)p * DMODEL + j] + bf[j];
    float f0 = sigmoidf_(xfb + fraw[(size_t)(2 * p) * DMODEL + j]);
    float f1 = sigmoidf_(xfb + fraw[(size_t)(2 * p + 1) * DMODEL + j]);
    size_t o0 = (size_t)c0 * DMODEL + j, o1 = (size_t)c1 * DMODEL + j;
    csum[idx] = f0 * c[o0] + f1 * c[o1];
    hsum[idx] = (__bf16)(h[o0] + h[o1]);
}

// ---------------------------------------------------------------------------
// LSTM cell update for the active nodes of one level.
// ---------------------------------------------------------------------------
__global__ void node_update_kernel(const float* __restrict__ xiou,  // nact x 768
                                   const float* __restrict__ biou,  // 768
                                   const float* __restrict__ iou2,  // nact x 768 or null
                                   const float* __restrict__ csum,  // nact x 256 or null
                                   const int*   __restrict__ act,   // nact
                                   float* __restrict__ h, float* __restrict__ c,
                                   __bf16* __restrict__ hb,
                                   float* __restrict__ root,        // nact x 256 or null
                                   int nact)
{
    int idx = blockIdx.x * blockDim.x + threadIdx.x;
    if (idx >= nact * DMODEL) return;
    int p = idx >> 8, j = idx & 255;
    size_t rp = (size_t)p * 768;
    float i_ = xiou[rp + j]       + biou[j];
    float o_ = xiou[rp + 256 + j] + biou[256 + j];
    float u_ = xiou[rp + 512 + j] + biou[512 + j];
    if (iou2) { i_ += iou2[rp + j]; o_ += iou2[rp + 256 + j]; u_ += iou2[rp + 512 + j]; }
    float cs = csum ? csum[idx] : 0.0f;
    float cn = sigmoidf_(i_) * tanhf(u_) + cs;
    float hn = sigmoidf_(o_) * tanhf(cn);
    size_t off = (size_t)act[p] * DMODEL + j;
    h[off] = hn; c[off] = cn; hb[off] = (__bf16)hn;
    if (root) root[idx] = hn;
}

// ---------------------------------------------------------------------------
// Head: s = <h_c,h_a>; logits = relu(fc2 @ relu(s*(fc1 @ h_b) + b1) + b2)
// ---------------------------------------------------------------------------
__global__ void head_kernel(const float* __restrict__ rc, const float* __restrict__ ra,
                            const float* __restrict__ rb,
                            const float* __restrict__ fc1w, const float* __restrict__ fc1b,
                            const float* __restrict__ fc2w, const float* __restrict__ fc2b,
                            float* __restrict__ out)
{
    __shared__ float shb[256];
    __shared__ float sred[128];
    __shared__ float shid[128];
    int b = blockIdx.x, t = threadIdx.x;   // 128 threads
    const float* pc = rc + (size_t)b * 256;
    const float* pa = ra + (size_t)b * 256;
    const float* pb = rb + (size_t)b * 256;
    float d = 0.0f;
    for (int k = t; k < 256; k += 128) { shb[k] = pb[k]; d += pc[k] * pa[k]; }
    sred[t] = d;
    __syncthreads();
    for (int s = 64; s > 0; s >>= 1) {
        if (t < s) sred[t] += sred[t + s];
        __syncthreads();
    }
    float s = sred[0];
    float acc = 0.0f;
    for (int k = 0; k < 256; ++k) acc += fc1w[(size_t)t * 256 + k] * shb[k];
    float hid = s * acc + fc1b[t];
    shid[t] = hid > 0.0f ? hid : 0.0f;
    __syncthreads();
    if (t < 3) {
        float a2 = 0.0f;
        for (int k = 0; k < 128; ++k) a2 += fc2w[(size_t)t * 128 + k] * shid[k];
        a2 += fc2b[t];
        out[(size_t)b * 3 + t] = a2 > 0.0f ? a2 : 0.0f;
    }
}

// ---------------------------------------------------------------------------
// Host side
// ---------------------------------------------------------------------------
static inline size_t alignup(size_t x) { return (x + 255) & ~(size_t)255; }

static void launch_gemm(const __bf16* A, const __bf16* W, float* out,
                        int M, int O, const int* g1, const int* g2, hipStream_t s)
{
    dim3 block(32, 4);                       // 4 waves/block
    int strips = M / 64;                     // M is always a multiple of 64
    dim3 grid(O / 16, (strips + 3) / 4);
    hipLaunchKernelGGL(wmma_gemm_kernel, grid, block, 0, s, A, W, out, M, O, 256, g1, g2);
}

extern "C" void kernel_launch(void* const* d_in, const int* in_sizes, int n_in,
                              void* d_out, int out_size, void* d_ws, size_t ws_size,
                              hipStream_t stream)
{
    const int*   feat[3] = { (const int*)d_in[0], (const int*)d_in[1], (const int*)d_in[2] };
    const float* emb  = (const float*)d_in[8];
    const float* Wiou = (const float*)d_in[9];
    const float* biou = (const float*)d_in[10];
    const float* Uiou = (const float*)d_in[11];
    const float* Wf   = (const float*)d_in[12];
    const float* bfv  = (const float*)d_in[13];
    const float* Uf   = (const float*)d_in[14];
    const float* fc1w = (const float*)d_in[15];
    const float* fc1b = (const float*)d_in[16];
    const float* fc2w = (const float*)d_in[17];
    const float* fc2b = (const float*)d_in[18];

    char* w = (char*)d_ws;
    size_t off = 0;
    auto carve = [&](size_t bytes) { void* p = w + off; off = alignup(off + bytes); return p; };

    float*  h      = (float*) carve((size_t)NNODES * DMODEL * 4);
    float*  c      = (float*) carve((size_t)NNODES * DMODEL * 4);
    __bf16* hb     = (__bf16*)carve((size_t)NNODES * DMODEL * 2);
    __bf16* embB   = (__bf16*)carve((size_t)10000 * 256 * 2);
    __bf16* wiouB  = (__bf16*)carve((size_t)768 * 256 * 2);
    __bf16* uiouB  = (__bf16*)carve((size_t)768 * 256 * 2);
    __bf16* wfB    = (__bf16*)carve((size_t)256 * 256 * 2);
    __bf16* ufB    = (__bf16*)carve((size_t)256 * 256 * 2);
    int*    act    = (int*)   carve((size_t)NNODES * 4);
    int*    child  = (int*)   carve((size_t)(NNODES - NTREE) * 4);
    float*  roots  = (float*) carve((size_t)3 * NTREE * DMODEL * 4);
    float*  xiou   = (float*) carve((size_t)16384 * 768 * 4);
    float*  iou2   = (float*) carve((size_t)8192 * 768 * 4);
    float*  fraw   = (float*) carve((size_t)16384 * 256 * 4);
    float*  xf     = (float*) carve((size_t)8192 * 256 * 4);
    float*  csum   = (float*) carve((size_t)8192 * 256 * 4);
    __bf16* hsum   = (__bf16*)carve((size_t)8192 * 256 * 2);

    // One-time staging: bf16 weights/embedding + heap index tables.
    auto conv = [&](const float* s, __bf16* d, int n) {
        hipLaunchKernelGGL(f32_to_bf16_kernel, dim3((n + 255) / 256), dim3(256), 0, stream, s, d, n);
    };
    conv(emb,  embB,  10000 * 256);
    conv(Wiou, wiouB, 768 * 256);
    conv(Uiou, uiouB, 768 * 256);
    conv(Wf,   wfB,   256 * 256);
    conv(Uf,   ufB,   256 * 256);
    hipLaunchKernelGGL(build_idx_kernel, dim3((NNODES + 255) / 256), dim3(256), 0, stream, act, child);

    for (int run = 0; run < 3; ++run) {
        const int* ft = feat[run];
        for (int lvl = 0; lvl < 8; ++lvl) {
            int dep  = 7 - lvl;
            int nact = NTREE << dep;
            const int* actp = act + NTREE * (256 - (1 << (dep + 1)));
            int ethreads = nact * DMODEL;
            dim3 eg((ethreads + 255) / 256), eb(256);

            if (lvl == 0) {
                launch_gemm(embB, wiouB, xiou, nact, 768, actp, ft, stream);
                hipLaunchKernelGGL(node_update_kernel, eg, eb, 0, stream,
                                   xiou, biou, (const float*)nullptr, (const float*)nullptr,
                                   actp, h, c, hb, (float*)nullptr, nact);
            } else {
                const int* chp = child + 2 * NTREE * (128 - (1 << (dep + 1)));
                launch_gemm(hb,   ufB,   fraw, 2 * nact, 256, chp,  nullptr, stream);
                launch_gemm(embB, wfB,   xf,   nact,     256, actp, ft,      stream);
                hipLaunchKernelGGL(edge_kernel, eg, eb, 0, stream,
                                   fraw, xf, bfv, h, c, chp, csum, hsum, nact);
                launch_gemm(hsum, uiouB, iou2, nact, 768, nullptr, nullptr, stream);
                launch_gemm(embB, wiouB, xiou, nact, 768, actp,    ft,      stream);
                float* rootp = (lvl == 7) ? roots + (size_t)run * NTREE * DMODEL : nullptr;
                hipLaunchKernelGGL(node_update_kernel, eg, eb, 0, stream,
                                   xiou, biou, iou2, csum, actp, h, c, hb, rootp, nact);
            }
        }
    }

    hipLaunchKernelGGL(head_kernel, dim3(NTREE), dim3(128), 0, stream,
                       roots, roots + (size_t)NTREE * DMODEL, roots + (size_t)2 * NTREE * DMODEL,
                       fc1w, fc1b, fc2w, fc2b, (float*)d_out);
}
